// SMoG_31550829756755
// MI455X (gfx1250) — matrix operand order repaired and checked by reading, not compiled, not run on hardware
//
#include <hip/hip_runtime.h>
#include <hip/hip_fp16.h>

typedef __attribute__((ext_vector_type(16))) _Float16 v16h;
typedef __attribute__((ext_vector_type(8)))  _Float16 v8h;
typedef __attribute__((ext_vector_type(8)))  float    v8f;

#define DIM     256
#define B_ROWS  16384
#define G_ROWS  8192
#define BETA_F  0.99f

// LDS B-tile: 16 groups x 256 halfs, row padded to 264 halfs (528 B) so the
// 16 lanes of a half-wave read disjoint bank groups on ds_load_b128.
#define LROW    264
#define NBUF    3

// ---------------------------------------------------------------------------
// Kernel 1: per-row L2 normalize (f32 in, f16 out). One 256-thread block/row.
// ---------------------------------------------------------------------------
__global__ __launch_bounds__(256)
void normalize_rows_f16(const float* __restrict__ src, _Float16* __restrict__ dst) {
  const int row = blockIdx.x;
  const int tid = threadIdx.x;                 // 0..255, one element each
  float v = src[row * DIM + tid];
  float s = v * v;
  #pragma unroll
  for (int m = 16; m >= 1; m >>= 1) s += __shfl_xor(s, m, 32);
  __shared__ float wsum[8];
  if ((tid & 31) == 0) wsum[tid >> 5] = s;
  __syncthreads();
  float tot = 0.f;
  #pragma unroll
  for (int i = 0; i < 8; ++i) tot += wsum[i];
  const float scale = 1.0f / fmaxf(sqrtf(tot), 1e-12f);   // matches F.normalize
  dst[row * DIM + tid] = (_Float16)(v * scale);
}

// ---------------------------------------------------------------------------
// Kernel 2: fused cosine-GEMM + per-row argmax via v_wmma_f32_16x16x32_f16.
//  * Each wave owns TWO 16-row A tiles (32 rows, K=256 resident: 128 VGPRs).
//  * B tiles (16 groups x 256 halfs = 8KB) are staged in LDS ONCE PER BLOCK
//    with global_load_async_to_lds_b128 (ASYNCcnt), triple-buffered and
//    software-pipelined: prefetch tile i+1 while WMMAs run on tile i.
//  * Each LDS B fragment feeds 2 WMMAs (independent acc chains).
// 128 threads = 4 waves per block; block covers 128 rows.
// ---------------------------------------------------------------------------
__global__ __launch_bounds__(128)
void assign_argmax_wmma(const _Float16* __restrict__ xn,
                        const _Float16* __restrict__ gn,
                        int* __restrict__ assign) {
  __shared__ _Float16 lbuf[NBUF][16 * LROW];

  const int lane    = threadIdx.x & 31;
  const int wave    = threadIdx.x >> 5;
  const int rowbase = blockIdx.x * 128 + wave * 32;   // rows rowbase..rowbase+31
  const int hi      = lane >> 4;      // which half-wave
  const int ln      = lane & 15;

  // --- A fragments: ISA 16-bit A 16x32 layout.
  // lanes 0-15:  M=ln, halfs = K {0..7, 16..23}; lanes 16-31: K {8..15, 24..31}
  v16h a0[8], a1[8];
  {
    const _Float16* ar0 = xn + (size_t)(rowbase + ln) * DIM + hi * 8;
    const _Float16* ar1 = xn + (size_t)(rowbase + 16 + ln) * DIM + hi * 8;
    #pragma unroll
    for (int k = 0; k < 8; ++k) {
      const _Float16* p0 = ar0 + k * 32;
      const _Float16* p1 = ar1 + k * 32;
      v8h lo0 = *(const v8h*)(p0);        // K = kb + hi*8 + 0..7
      v8h hh0 = *(const v8h*)(p0 + 16);   // K = kb + 16 + hi*8 + 0..7
      v8h lo1 = *(const v8h*)(p1);
      v8h hh1 = *(const v8h*)(p1 + 16);
      v16h a, b;
      #pragma unroll
      for (int i = 0; i < 8; ++i) { a[i] = lo0[i]; a[i + 8] = hh0[i];
                                    b[i] = lo1[i]; b[i + 8] = hh1[i]; }
      a0[k] = a;
      a1[k] = b;
    }
  }

  // Async copy of one 8KB B tile into lbuf[buf]: 512 x b128 chunks, 4/thread.
  // Chunk c: tile row = c>>5, 16B column group = c&31 (row stride 528B in LDS).
  auto issue_tile = [&](int tile, int buf) {
    const char* gbase = (const char*)gn + (size_t)tile * (16 * DIM * 2);
    unsigned lbase = (unsigned)(size_t)(&lbuf[buf][0]);   // LDS byte offset
    #pragma unroll
    for (int r = 0; r < 4; ++r) {
      int c = threadIdx.x + r * 128;
      unsigned loff = lbase + (unsigned)((c >> 5) * (LROW * 2) + (c & 31) * 16);
      unsigned long long gaddr = (unsigned long long)(size_t)(gbase + c * 16);
      asm volatile("global_load_async_to_lds_b128 %0, %1, off"
                   :: "v"(loff), "v"(gaddr) : "memory");
    }
  };

  float runmax0[8], runmax1[8];
  int   runidx0[8], runidx1[8];
  #pragma unroll
  for (int r = 0; r < 8; ++r) {
    runmax0[r] = -3.4e38f; runidx0[r] = 0;
    runmax1[r] = -3.4e38f; runidx1[r] = 0;
  }

  const int NT = G_ROWS / 16;          // 512 N-tiles
  issue_tile(0, 0);                    // prologue

  for (int gt = 0; gt < NT; ++gt) {
    const int buf = gt % NBUF;
    // Prefetch next tile, then wait for THIS tile's 4 copies (in-order
    // completion => asynccnt<=4 leaves only the prefetch outstanding).
    if (gt + 1 < NT) {
      issue_tile(gt + 1, (gt + 1) % NBUF);
      asm volatile("s_wait_asynccnt 0x4" ::: "memory");
    } else {
      asm volatile("s_wait_asynccnt 0x0" ::: "memory");
    }
    __syncthreads();   // all waves' portions of tile gt are in LDS

    // --- B fragment from LDS: lanes 0-15: N=ln, K {0..7, 16..23}; lanes 16-31: K {8..15, 24..31}.
    const _Float16* lb = &lbuf[buf][ln * LROW + hi * 8];
    v8f acc0 = {0.f, 0.f, 0.f, 0.f, 0.f, 0.f, 0.f, 0.f};
    v8f acc1 = {0.f, 0.f, 0.f, 0.f, 0.f, 0.f, 0.f, 0.f};
    #pragma unroll
    for (int k = 0; k < 8; ++k) {
      v8h lo = *(const v8h*)(lb + k * 32);      // K = kb + hi*8 + 0..7
      v8h hh = *(const v8h*)(lb + k * 32 + 16); // K = kb + 16 + hi*8 + 0..7
      v16h b;
      #pragma unroll
      for (int i = 0; i < 8; ++i) { b[i] = lo[i]; b[i + 8] = hh[i]; }
      acc0 = __builtin_amdgcn_wmma_f32_16x16x32_f16(
          false, a0[k], false, b, (short)0, acc0, false, false);
      acc1 = __builtin_amdgcn_wmma_f32_16x16x32_f16(
          false, a1[k], false, b, (short)0, acc1, false, false);
    }
    // C layout: vgpr r, lane l -> row = r + 8*hi, col = ln. Running argmax;
    // strict '>' keeps first occurrence (columns visited in ascending order).
    const int col = gt * 16 + ln;
    #pragma unroll
    for (int r = 0; r < 8; ++r) {
      if (acc0[r] > runmax0[r]) { runmax0[r] = acc0[r]; runidx0[r] = col; }
      if (acc1[r] > runmax1[r]) { runmax1[r] = acc1[r]; runidx1[r] = col; }
    }
  }

  // Cross-lane reduction over the 16 columns per half-wave (masks < 16 stay
  // inside each half). Tie-break toward lower index (argmax semantics).
  #pragma unroll
  for (int r = 0; r < 8; ++r) {
    float m0 = runmax0[r]; int i0 = runidx0[r];
    float m1 = runmax1[r]; int i1 = runidx1[r];
    #pragma unroll
    for (int s = 1; s < 16; s <<= 1) {
      float om0 = __shfl_xor(m0, s, 32);
      int   oi0 = __shfl_xor(i0, s, 32);
      float om1 = __shfl_xor(m1, s, 32);
      int   oi1 = __shfl_xor(i1, s, 32);
      if (om0 > m0 || (om0 == m0 && oi0 < i0)) { m0 = om0; i0 = oi0; }
      if (om1 > m1 || (om1 == m1 && oi1 < i1)) { m1 = om1; i1 = oi1; }
    }
    runidx0[r] = i0;
    runidx1[r] = i1;
  }
  if (ln == 0) {
    #pragma unroll
    for (int r = 0; r < 8; ++r) {
      assign[rowbase + hi * 8 + r]      = runidx0[r];
      assign[rowbase + 16 + hi * 8 + r] = runidx1[r];
    }
  }
}

// ---------------------------------------------------------------------------
// Kernel 3: zero scratch (counts + sums, contiguous).
// ---------------------------------------------------------------------------
__global__ void zero_f32(float* __restrict__ p, int n) {
  int i = blockIdx.x * 256 + threadIdx.x;
  if (i < n) p[i] = 0.f;
}

// ---------------------------------------------------------------------------
// Kernel 4: segment-sum scatter with f32 global atomics (L2-resident).
// ---------------------------------------------------------------------------
__global__ __launch_bounds__(256)
void scatter_sums(const float* __restrict__ x, const int* __restrict__ assign,
                  float* __restrict__ sums, float* __restrict__ counts) {
  const int row = blockIdx.x;
  const int d   = threadIdx.x;
  const int a   = assign[row];
  atomicAdd(&sums[(size_t)a * DIM + d], x[(size_t)row * DIM + d]);
  if (d == 0) atomicAdd(&counts[a], 1.0f);
}

// ---------------------------------------------------------------------------
// Kernel 5: EMA blend: out = beta*g + (1-beta)*sums/max(counts,1)
// ---------------------------------------------------------------------------
__global__ __launch_bounds__(256)
void ema_update(const float* __restrict__ g, const float* __restrict__ sums,
                const float* __restrict__ counts, float* __restrict__ out) {
  const int gi = blockIdx.x;
  const int d  = threadIdx.x;
  const float c = fmaxf(counts[gi], 1.0f);
  out[(size_t)gi * DIM + d] =
      BETA_F * g[(size_t)gi * DIM + d] + (1.0f - BETA_F) * (sums[(size_t)gi * DIM + d] / c);
}

// ---------------------------------------------------------------------------
extern "C" void kernel_launch(void* const* d_in, const int* in_sizes, int n_in,
                              void* d_out, int out_size, void* d_ws, size_t ws_size,
                              hipStream_t stream) {
  const float* x = (const float*)d_in[0];       // [16384, 256] f32
  const float* g = (const float*)d_in[1];       // [8192, 256] f32
  float* out = (float*)d_out;                   // [8192, 256] f32

  // Workspace layout (~21 MB total)
  char* ws = (char*)d_ws;
  _Float16* xn = (_Float16*)ws;  ws += (size_t)B_ROWS * DIM * sizeof(_Float16);
  _Float16* gn = (_Float16*)ws;  ws += (size_t)G_ROWS * DIM * sizeof(_Float16);
  int* assign  = (int*)ws;       ws += (size_t)B_ROWS * sizeof(int);
  float* counts = (float*)ws;    ws += (size_t)G_ROWS * sizeof(float);
  float* sums   = (float*)ws;    // [G_ROWS * DIM], contiguous after counts

  // 1) normalize to f16
  normalize_rows_f16<<<B_ROWS, 256, 0, stream>>>(x, xn);
  normalize_rows_f16<<<G_ROWS, 256, 0, stream>>>(g, gn);

  // 2) fused WMMA GEMM + argmax (each block = 4 waves x 32 rows = 128 rows)
  assign_argmax_wmma<<<B_ROWS / 128, 128, 0, stream>>>(xn, gn, assign);

  // 3) zero counts+sums (contiguous region)
  const int nz = G_ROWS + G_ROWS * DIM;
  zero_f32<<<(nz + 255) / 256, 256, 0, stream>>>(counts, nz);

  // 4) scatter segment sums
  scatter_sums<<<B_ROWS, 256, 0, stream>>>(x, assign, sums, counts);

  // 5) EMA blend into output
  ema_update<<<G_ROWS, 256, 0, stream>>>(g, sums, counts, out);
}
